// DPSpikingDecoder_86517821216469
// MI455X (gfx1250) — matrix operand order repaired and checked by reading, not compiled, not run on hardware
//
#include <hip/hip_runtime.h>

typedef __attribute__((ext_vector_type(16))) _Float16 v16h;
typedef __attribute__((ext_vector_type(8)))  float    v8f;

#define B_  8
#define C_  32
#define T_  960
#define F_  256
#define W_  40      // num windows
#define LDP 24
#define NDP 12
#define D_  (W_ * F_)   // 10240
#define H_  20          // hidden = W_/2

#define NITER   (D_ / 32)        // 320 WMMA K-iterations total
#define KSL     8                // K-slices (waves per N-tile)
#define ITERS_W (NITER / KSL)    // 40 iterations per wave
#define AFRAG_N (NITER * 32 * 16)        // halves in Apack
#define BFRAG_N (2 * NITER * 32 * 16)    // halves in Bpack

// ---------------------------------------------------------------------------
// Kernel 1: fused channel-sum + banded window reduction.
// One block per (b, w); 256 threads = 4 channel-groups x 64 f4-groups.
// Reads the 252MB spikes tensor exactly once with 128b coalesced loads.
// ---------------------------------------------------------------------------
__global__ __launch_bounds__(256) void k_pool(const float* __restrict__ spikes,
                                              float* __restrict__ Lb,
                                              float* __restrict__ Eb) {
    const int bw = blockIdx.x;
    const int b = bw / W_, w = bw % W_;
    const int tid = threadIdx.x;
    const int cg = tid >> 6;      // channel group 0..3 (8 channels each)
    const int fi = tid & 63;      // f4 group
    const int f0 = fi * 4;

    // Weight tables: fully unrolled from literals -> constant-folded.
    float hq[LDP], eq[LDP];
    {
        float hacc = 0.f;
#pragma unroll
        for (int q = LDP - 1; q >= 0; --q) {
            const float a = (q >= NDP) ? (1.0f / 12.0f) : (-1.0f / 12.0f);
            hacc = fmaf(0.5f, hacc, a);   // h(q) = a(q) + 0.5*h(q+1)
            hq[q] = hacc;
        }
        float e = 1.f;
#pragma unroll
        for (int q = LDP - 1; q >= 0; --q) { eq[q] = e; e *= 0.5f; } // 2^(q-23)
    }

    const size_t TF = (size_t)T_ * F_;
    const float* base = spikes + ((size_t)b * C_ + (size_t)cg * 8) * TF
                               + (size_t)(w * LDP) * F_ + f0;

    float L[4] = {0.f, 0.f, 0.f, 0.f};
    float E[4] = {0.f, 0.f, 0.f, 0.f};
    for (int cc = 0; cc < 8; ++cc) {
        const float* p = base + (size_t)cc * TF;
#pragma unroll
        for (int q = 0; q < LDP; ++q) {
            const float4 x = *(const float4*)(p + q * F_);
            const float hv = hq[q], ev = eq[q];
            L[0] = fmaf(hv, x.x, L[0]); E[0] = fmaf(ev, x.x, E[0]);
            L[1] = fmaf(hv, x.y, L[1]); E[1] = fmaf(ev, x.y, E[1]);
            L[2] = fmaf(hv, x.z, L[2]); E[2] = fmaf(ev, x.z, E[2]);
            L[3] = fmaf(hv, x.w, L[3]); E[3] = fmaf(ev, x.w, E[3]);
        }
    }

    __shared__ float sL[4][64][4];
    __shared__ float sE[4][64][4];
#pragma unroll
    for (int i = 0; i < 4; ++i) { sL[cg][fi][i] = L[i]; sE[cg][fi][i] = E[i]; }
    __syncthreads();
    if (cg == 0) {
#pragma unroll
        for (int i = 0; i < 4; ++i) {
            const float l = sL[0][fi][i] + sL[1][fi][i] + sL[2][fi][i] + sL[3][fi][i];
            const float e = sE[0][fi][i] + sE[1][fi][i] + sE[2][fi][i] + sE[3][fi][i];
            const size_t o = ((size_t)b * W_ + w) * F_ + f0 + i;
            Lb[o] = l * (1.0f / C_);   // fold channel mean here
            Eb[o] = e * (1.0f / C_);
        }
    }
}

// ---------------------------------------------------------------------------
// Kernel 2: exact 40-step window-level carry recurrence per (b,f) column.
// dp[w] = C1*carry + L[w];  carry' = 2^-24*carry + E[w]
// ---------------------------------------------------------------------------
__global__ __launch_bounds__(256) void k_scan(const float* __restrict__ Lb,
                                              const float* __restrict__ Eb,
                                              float* __restrict__ dp) {
    const int b = blockIdx.x;
    const int f = threadIdx.x;
    const float C1f = -(1.0f / 12.0f) * (1.0f - 0x1p-12f) * (1.0f - 0x1p-12f);
    const float D24 = 0x1p-24f;
    float carry = 0.f;
    for (int w = 0; w < W_; ++w) {
        const size_t o = ((size_t)b * W_ + w) * F_ + f;
        dp[o] = fmaf(C1f, carry, Lb[o]);
        carry = fmaf(D24, carry, Eb[o]);
    }
}

// ---------------------------------------------------------------------------
// Kernel 3a: pack dp (zero-pad rows 8..15) and W1 (zero-pad cols 20..31) into
// the exact per-lane WMMA fragment layout (f16), so the WMMA loop has zero
// divergence and pure vector loads.
//   Apack[(iter*32+lane)*16+e]             A 16x32 f16 fragment layout
//   Bpack[((ntile*NITER+iter)*32+lane)*16+e]  B 32x16 f16 fragment layout
// ---------------------------------------------------------------------------
__global__ __launch_bounds__(256) void k_pack(const float* __restrict__ dp,
                                              const float* __restrict__ W1,
                                              _Float16* __restrict__ Apack,
                                              _Float16* __restrict__ Bpack) {
    const int idx = blockIdx.x * 256 + threadIdx.x;
    if (idx < AFRAG_N) {
        const int e    = idx & 15;
        const int lane = (idx >> 4) & 31;
        const int iter = idx >> 9;
        const int row  = lane & 15;
        const int kgrp = lane >> 4;
        // A 16x32 f16 layout: lanes0-15 K in {0..7,16..23}, lanes16-31 +8
        const int ka = iter * 32 + ((e < 8) ? e : (e + 8)) + 8 * kgrp;
        const float v = (row < B_) ? dp[(size_t)row * D_ + ka] : 0.f;
        Apack[idx] = (_Float16)v;
    } else {
        const int j2   = idx - AFRAG_N;
        const int e    = j2 & 15;
        const int lane = (j2 >> 4) & 31;
        const int g    = j2 >> 9;            // 0 .. 2*NITER-1
        const int iter = g % NITER;
        const int nt   = g / NITER;
        const int kgrp = lane >> 4;
        const int jcol = (lane & 15) + 16 * nt;
        // B 32x16 f16 layout: lanes0-15 K in {0..7,16..23}, lanes16-31 +8
        const int kb = iter * 32 + ((e < 8) ? e : (e + 8)) + 8 * kgrp;
        const float v = (jcol < H_) ? W1[(size_t)kb * H_ + jcol] : 0.f;
        Bpack[j2] = (_Float16)v;
    }
}

// ---------------------------------------------------------------------------
// Kernel 3b: GEMM1 via v_wmma_f32_16x16x32_f16 (16 waves = 8 K-slices x 2
// N-tiles, 40 iters each), LDS reduce + relu, then 20x40 head + softmax.
// WMMA loop: uniform trip count, unconditional b128 loads, EXEC all-ones.
// ---------------------------------------------------------------------------
__global__ __launch_bounds__(512) void k_head(const _Float16* __restrict__ Apack,
                                              const _Float16* __restrict__ Bpack,
                                              const float* __restrict__ b1,
                                              const float* __restrict__ W2,
                                              const float* __restrict__ b2,
                                              float* __restrict__ s_out) {
    const int tid = threadIdx.x;
    const int lane = tid & 31;
    const int wid = tid >> 5;        // 0..15
    const int ntile = wid & 1;       // N tile: j 0..15 / 16..31
    const int kslice = wid >> 1;     // 0..7

    __shared__ float part[KSL][2][8][16];   // [kslice][ntile][row(b)][laneN]
    __shared__ float hsh[B_][H_];

    const v16h* ap = (const v16h*)Apack + (size_t)(kslice * ITERS_W) * 32 + lane;
    const v16h* bp = (const v16h*)Bpack
                   + ((size_t)ntile * NITER + (size_t)kslice * ITERS_W) * 32 + lane;

    v8f acc = {};
#pragma unroll 4
    for (int it = 0; it < ITERS_W; ++it) {
        const v16h a  = ap[it * 32];
        const v16h bm = bp[it * 32];
        acc = __builtin_amdgcn_wmma_f32_16x16x32_f16(
            false, a, false, bm, (short)0, acc, false, false);
    }

    // D layout: lanes 0-15 hold N=lane, batch row r in acc[r] (r=0..7)
    if (lane < 16) {
#pragma unroll
        for (int r = 0; r < 8; ++r) part[kslice][ntile][r][lane] = acc[r];
    }
    __syncthreads();

    if (tid < B_ * H_) {
        const int b = tid / H_, j = tid % H_;
        float sum = b1[j];
        const int nt = j >> 4, ln = j & 15;
#pragma unroll
        for (int ks = 0; ks < KSL; ++ks) sum += part[ks][nt][b][ln];
        hsh[b][j] = sum > 0.f ? sum : 0.f;   // relu
    }
    __syncthreads();

    if (tid < B_) {
        const int b = tid;
        float z[W_];
        float zmax = -3.4e38f;
        for (int k = 0; k < W_; ++k) {
            float a2 = b2[k];
            for (int j = 0; j < H_; ++j) a2 = fmaf(hsh[b][j], W2[j * W_ + k], a2);
            z[k] = a2;
            zmax = fmaxf(zmax, a2);
        }
        float zsum = 0.f;
        for (int k = 0; k < W_; ++k) { z[k] = __expf(z[k] - zmax); zsum += z[k]; }
        const float inv = 1.0f / zsum;
        for (int k = 0; k < W_; ++k) s_out[b * W_ + k] = z[k] * inv;
    }
}

// ---------------------------------------------------------------------------
// Kernel 4: out[b,w,f] = dp[b,w,f] * s[b,w]
// ---------------------------------------------------------------------------
__global__ __launch_bounds__(256) void k_scale(const float* __restrict__ dp,
                                               const float* __restrict__ s,
                                               float* __restrict__ out) {
    const int i = blockIdx.x * 256 + threadIdx.x;
    const int b = i / D_;
    const int w = (i % D_) / F_;
    out[i] = dp[i] * s[b * W_ + w];
}

extern "C" void kernel_launch(void* const* d_in, const int* in_sizes, int n_in,
                              void* d_out, int out_size, void* d_ws, size_t ws_size,
                              hipStream_t stream) {
    (void)in_sizes; (void)n_in; (void)out_size; (void)ws_size;
    const float* spikes = (const float*)d_in[0];
    const float* W1 = (const float*)d_in[1];
    const float* b1 = (const float*)d_in[2];
    const float* W2 = (const float*)d_in[3];
    const float* b2 = (const float*)d_in[4];

    float* ws = (float*)d_ws;
    float* Lb = ws;                       // 81920 floats
    float* Eb = ws + 81920;               // 81920 floats
    float* dp = ws + 2 * 81920;           // 81920 floats
    float* sA = ws + 3 * 81920;           // 320 floats (padded to 512)
    // f16 fragment buffers, 32B-aligned (float region is 32B-divisible)
    _Float16* Apack = (_Float16*)(ws + 3 * 81920 + 512);   // AFRAG_N halves
    _Float16* Bpack = Apack + AFRAG_N;                     // BFRAG_N halves

    hipLaunchKernelGGL(k_pool,  dim3(B_ * W_), dim3(256), 0, stream, spikes, Lb, Eb);
    hipLaunchKernelGGL(k_scan,  dim3(B_),      dim3(256), 0, stream, Lb, Eb, dp);
    hipLaunchKernelGGL(k_pack,  dim3((AFRAG_N + BFRAG_N) / 256), dim3(256), 0, stream,
                       dp, W1, Apack, Bpack);
    hipLaunchKernelGGL(k_head,  dim3(1), dim3(512), 0, stream,
                       Apack, Bpack, b1, W2, b2, sA);
    hipLaunchKernelGGL(k_scale, dim3((B_ * D_) / 256), dim3(256), 0, stream,
                       dp, sA, (float*)d_out);
}
